// GLA2D_74105365725893
// MI455X (gfx1250) — matrix operand order repaired and checked by reading, not compiled, not run on hardware
//
#include <hip/hip_runtime.h>
#include <math.h>

// ---------------------------------------------------------------------------
// GLA-2D forward for MI455X (gfx1250, wave32, WMMA).
// bf16 operands + f32 accumulation (v_wmma_f32_16x16x32_bf16); GLA recurrent
// state S stays f32 in LDS. All LDS WMMA operands are stored so fragment
// loads are contiguous (ds_load_b128) with conflict-free padded strides.
// ---------------------------------------------------------------------------

typedef __attribute__((ext_vector_type(16))) __bf16 v16bf;
typedef __attribute__((ext_vector_type(8)))  float  v8f;

#define NSEQ   3136
#define BATCH  8
#define DMODEL 512
#define KEYD   512
#define VALD   1024
#define NH     4
#define HQK    128
#define HV     256
#define CHUNK  64
#define NCHUNK 49
#define MROWS  (BATCH * NSEQ)   // 25088 rows
#define NFUSE  3088             // 2048 qkv + 1024 g + 16 gk1
#define QKSCALE 0.08838834764831845f   // 128^-0.5

// Padded LDS row strides (bf16 elems): stride*2 bytes is a multiple of 16
// (keeps b128-capable alignment) and stride/8 is odd so the 16 fragment rows
// hit 16 disjoint 4-bank groups (conflict-free ds_load_b128).
#define LDQ 136   // for 128-wide rows (qe, kd, Sb^T)
#define LDT 72    // for 64-wide rows (kt^T, v^T, A)

// --------------------------- WMMA helpers ----------------------------------

__device__ __forceinline__ v8f wmma_bf16(v16bf a, v16bf b, v8f c) {
  // D = A(16x32) * B(32x16) + C, f32 accumulate
  return __builtin_amdgcn_wmma_f32_16x16x32_bf16(false, a, false, b,
                                                 (short)0, c, false, false);
}

// A fragment (16x32 bf16) from row-major [M,K]; base -> (row0, k0).
// Lanes 0-15: row = lane, K {0..7, 16..23}; lanes 16-31: K {8..15, 24..31}.
__device__ __forceinline__ v16bf frag_a_rm(const __bf16* base, int ld, int lane) {
  const int m = lane & 15, g = (lane >> 4) << 3;
  const __bf16* p = base + (size_t)m * ld + g;
  v16bf f;
#pragma unroll
  for (int i = 0; i < 8; ++i) { f[i] = p[i]; f[i + 8] = p[i + 16]; }
  return f;
}

// B fragment (32x16) where B = W^T of row-major W[N,K]; base -> W(n0, k0).
// Lane n = col; lanes 0-15 carry K {0..7, 16..23}, lanes 16-31 carry K {8..15, 24..31}.
__device__ __forceinline__ v16bf frag_b_nk(const __bf16* base, int ld, int lane) {
  const int n = lane & 15, g = (lane >> 4) << 3;
  const __bf16* p = base + (size_t)n * ld + g;
  v16bf f;
#pragma unroll
  for (int i = 0; i < 8; ++i) { f[i] = p[i]; f[i + 8] = p[i + 16]; }
  return f;
}

// --------------------------- weight packing --------------------------------

__global__ __launch_bounds__(256)
void pack_weights_kernel(const float* __restrict__ qkv_w, const float* __restrict__ g_w,
                         const float* __restrict__ gk_w1, const float* __restrict__ o_w,
                         __bf16* __restrict__ W1, __bf16* __restrict__ W2) {
  const int n1 = 2048 * 512, n2 = 3072 * 512, n3 = NFUSE * 512;
  const int n4 = n3 + 512 * 1024;
  int idx = blockIdx.x * 256 + threadIdx.x;
  if (idx < n1)      W1[idx] = (__bf16)qkv_w[idx];
  else if (idx < n2) W1[idx] = (__bf16)g_w[idx - n1];
  else if (idx < n3) W1[idx] = (__bf16)gk_w1[idx - n2];
  else if (idx < n4) W2[idx - n3] = (__bf16)o_w[idx - n3];
}

// --------------------- depthwise 3x3 conv + SiLU ---------------------------

__global__ __launch_bounds__(256)
void conv_silu_kernel(const float* __restrict__ x, const float* __restrict__ w,
                      __bf16* __restrict__ xs) {
  size_t idx = (size_t)blockIdx.x * 256 + threadIdx.x;   // [b, t, c] linear
  const int c = (int)(idx & (DMODEL - 1));
  const int t = (int)((idx >> 9) % NSEQ);
  const int b = (int)(idx / ((size_t)NSEQ * DMODEL));
  const int y = t / 56, xx = t % 56;
  float s = 0.f;
#pragma unroll
  for (int dy = -1; dy <= 1; ++dy) {
    const int yy = y + dy;
    if (yy < 0 || yy >= 56) continue;
#pragma unroll
    for (int dx = -1; dx <= 1; ++dx) {
      const int xq = xx + dx;
      if (xq < 0 || xq >= 56) continue;
      s += x[((size_t)b * NSEQ + yy * 56 + xq) * DMODEL + c] *
           w[c * 9 + (dy + 1) * 3 + (dx + 1)];
    }
  }
  const float o = s / (1.f + __expf(-s));   // SiLU
  xs[idx] = (__bf16)o;
}

// ------------------------- generic WMMA GEMM -------------------------------
// C[M,Nn] = A[M,K] @ W[Nn,K]^T.  8 waves/block, each wave a 32x64 tile
// (block tile 64x256; 8 WMMAs per K-step). Per-lane pointers hoisted.
// MODE 0: fused epilogue scatters into qkv(bf16) / g(f32) / gk1(f32).
// MODE 1: plain f32 store (final projection into d_out).

template <int MODE>
__global__ __launch_bounds__(256)
void wmma_gemm_kernel(const __bf16* __restrict__ A, const __bf16* __restrict__ W,
                      int M, int Nn, int K,
                      __bf16* __restrict__ out_qkv, float* __restrict__ out_g,
                      float* __restrict__ out_gk1, float* __restrict__ out_f32) {
  const int lane = threadIdx.x & 31;
  const int wave = threadIdx.x >> 5;
  const int m0 = blockIdx.y * 64 + (wave >> 2) * 32;
  const int n0 = blockIdx.x * 256 + (wave & 3) * 64;
  if (m0 >= M) return;

  const int lm = lane & 15, lg8 = (lane >> 4) << 3, lg16 = (lane >> 4) << 4;
  const __bf16* aP0 = A + (size_t)(m0 + lm) * K + lg8;
  const __bf16* aP1 = aP0 + (size_t)16 * K;
  const __bf16* wP[4];
#pragma unroll
  for (int j = 0; j < 4; ++j) {
    int n = n0 + j * 16 + lm;
    n = (n < Nn) ? n : (Nn - 1);           // clamp ragged tail rows
    wP[j] = W + (size_t)n * K + lg8;
  }

  v8f acc[2][4] = {};
  for (int k0 = 0; k0 < K; k0 += 32) {
    __builtin_prefetch(aP0 + k0 + 128, 0, 1);   // global_prefetch_b8
    v16bf a0, a1;
#pragma unroll
    for (int i = 0; i < 8; ++i) {
      a0[i] = aP0[k0 + i]; a0[i + 8] = aP0[k0 + 16 + i];
      a1[i] = aP1[k0 + i]; a1[i + 8] = aP1[k0 + 16 + i];
    }
#pragma unroll
    for (int j = 0; j < 4; ++j) {
      v16bf bj;
#pragma unroll
      for (int i = 0; i < 8; ++i) { bj[i] = wP[j][k0 + i]; bj[i + 8] = wP[j][k0 + 16 + i]; }
      acc[0][j] = wmma_bf16(a0, bj, acc[0][j]);
      acc[1][j] = wmma_bf16(a1, bj, acc[1][j]);
    }
  }

  const int mlane = lg8;   // C/D layout: M = vgpr + 8*(lane>=16)
#pragma unroll
  for (int i = 0; i < 2; ++i)
#pragma unroll
    for (int j = 0; j < 4; ++j)
#pragma unroll
      for (int r = 0; r < 8; ++r) {
        const int m = m0 + i * 16 + r + mlane;
        const int n = n0 + j * 16 + lm;
        if (n >= Nn) continue;
        const float v = acc[i][j][r];
        if (MODE == 0) {
          if (n < 2048)      out_qkv[(size_t)m * 2048 + n] = (__bf16)v;
          else if (n < 3072) out_g[(size_t)m * 1024 + (n - 2048)] = v;
          else               out_gk1[(size_t)m * 16 + (n - 3072)] = v;
        } else {
          out_f32[(size_t)m * Nn + n] = v;
        }
      }
}

// ---------------------- gate expansion + log-sigmoid -----------------------

__global__ __launch_bounds__(256)
void gk_kernel(const float* __restrict__ gk1, const float* __restrict__ gk_w2,
               const float* __restrict__ gk_b2, float* __restrict__ gk_out) {
  size_t idx = (size_t)blockIdx.x * 256 + threadIdx.x;
  const int m = (int)(idx >> 10);
  const int j = (int)(idx & 1023);
  float s = gk_b2[j];
#pragma unroll
  for (int i = 0; i < 16; ++i) s += gk1[(size_t)m * 16 + i] * gk_w2[j * 16 + i];
  const float ls = fminf(s, 0.f) - log1pf(__expf(-fabsf(s)));  // log_sigmoid
  gk_out[idx] = ls * (1.f / 16.f);
}

// --------------------------- chunked GLA scan ------------------------------
// One block per (batch, head, direction, 64-wide dv slice): 256 blocks,
// 8 waves. State S[128][64] lives in LDS (f32). kt, v, and the S snapshot
// are stored TRANSPOSED in LDS so every WMMA operand is a contiguous padded
// row -> conflict-free ds_load_b128 fragments.

__global__ __launch_bounds__(256)
void gla_kernel(const __bf16* __restrict__ qkv, const float* __restrict__ gk,
                float* __restrict__ o_f, float* __restrict__ o_b) {
  extern __shared__ char smem_raw[];
  float*  S    = (float*)smem_raw;               // [128][64] f32 state (dk-major)
  __bf16* SbT  = (__bf16*)(S + 128 * 64);        // [64][LDQ] bf16 snapshot^T (dv-major)
  float*  gbuf = (float*)(SbT + 64 * LDQ);       // [64][128] gate chunk
  __bf16* qe   = (__bf16*)(gbuf + 64 * 128);     // [64][LDQ]
  __bf16* kd   = qe + 64 * LDQ;                  // [64][LDQ]
  __bf16* ktT  = kd + 64 * LDQ;                  // [128][LDT] kt^T (dk-major)
  __bf16* vbT  = ktT + 128 * LDT;                // [64][LDT]  v^T  (dv-major)
  __bf16* Ab   = vbT + 64 * LDT;                 // [64][LDT]  masked attn
  float*  btot = (float*)(Ab + 64 * LDT);        // [128]
  float*  ebt  = btot + 128;                     // [128] exp(btot)

  int bid = blockIdx.x;
  const int dvb = bid & 3;  bid >>= 2;
  const int dir = bid & 1;  bid >>= 1;
  const int h   = bid & 3;  bid >>= 2;
  const int b   = bid;

  const int tid = threadIdx.x, lane = tid & 31, wave = tid >> 5;
  const int qoff = h * HQK;
  const int koff = KEYD + h * HQK;
  const int voff = 2 * KEYD + h * HV + dvb * 64;
  const int goff = dir * KEYD + h * HQK;
  float* od = dir ? o_b : o_f;

  for (int i = tid; i < 128 * 64; i += 256) S[i] = 0.f;
  __syncthreads();

  for (int cc = 0; cc < NCHUNK; ++cc) {
    // ---- stage gate chunk (float4; sequence flip for backward pass) ----
    for (int i4 = tid; i4 < (CHUNK * HQK) / 4; i4 += 256) {
      const int t = i4 >> 5, c4 = (i4 & 31) << 2;
      const int row = dir ? (NSEQ - 1 - (cc * CHUNK + t)) : (cc * CHUNK + t);
      const float4 g4 =
          *(const float4*)&gk[((size_t)(b * NSEQ) + row) * VALD + goff + c4];
      *(float4*)&gbuf[t * HQK + c4] = g4;
    }
    __syncthreads();

    // ---- per-column cumsum totals ----
    if (tid < HQK) {
      float s = 0.f;
      for (int t = 0; t < CHUNK; ++t) s += gbuf[t * HQK + tid];
      btot[tid] = s;  ebt[tid] = __expf(s);
    }
    __syncthreads();

    // ---- build qe / kd / kt^T (threads 0..127) ; stage v^T (128..255) ----
    if (tid < HQK) {
      const int cidx = tid;
      const float bt = btot[cidx];
      float run = 0.f;
      for (int t = 0; t < CHUNK; ++t) {
        run += gbuf[t * HQK + cidx];
        const int row = dir ? (NSEQ - 1 - (cc * CHUNK + t)) : (cc * CHUNK + t);
        const size_t rb = ((size_t)(b * NSEQ) + row) * 2048;
        const float qv = (float)qkv[rb + qoff + cidx];
        const float kv = (float)qkv[rb + koff + cidx];
        qe[t * LDQ + cidx]  = (__bf16)(qv * __expf(run) * QKSCALE);
        kd[t * LDQ + cidx]  = (__bf16)(kv * __expf(-run));
        ktT[cidx * LDT + t] = (__bf16)(kv * __expf(bt - run));
      }
    } else {
      for (int i = tid - 128; i < CHUNK * 64; i += 128) {
        const int d = i & 63, t = i >> 6;   // d fastest: coalesced global read
        const int row = dir ? (NSEQ - 1 - (cc * CHUNK + t)) : (cc * CHUNK + t);
        vbT[d * LDT + t] = qkv[((size_t)(b * NSEQ) + row) * 2048 + voff + d];
      }
    }
    __syncthreads();

    // ---- A = tril(qe @ kd^T): 16 tiles over 8 waves ----
    for (int ti = wave * 2; ti < wave * 2 + 2; ++ti) {
      const int rt = ti >> 2, ct = ti & 3;
      if (rt < ct) {
#pragma unroll
        for (int r = 0; r < 8; ++r) {
          const int m = rt * 16 + r + ((lane >> 4) << 3);
          const int n = ct * 16 + (lane & 15);
          Ab[m * LDT + n] = (__bf16)0.f;
        }
      } else {
        v8f acc = {};
#pragma unroll
        for (int k0 = 0; k0 < HQK; k0 += 32) {
          v16bf a  = frag_a_rm(qe + rt * 16 * LDQ + k0, LDQ, lane);
          v16bf bb = frag_b_nk(kd + ct * 16 * LDQ + k0, LDQ, lane);
          acc = wmma_bf16(a, bb, acc);
        }
#pragma unroll
        for (int r = 0; r < 8; ++r) {
          const int tg = rt * 16 + r + ((lane >> 4) << 3);
          const int sg = ct * 16 + (lane & 15);
          Ab[tg * LDT + sg] = (__bf16)((tg >= sg) ? acc[r] : 0.f);
        }
      }
    }
    __syncthreads();

    // ---- snapshot old state, transposed, for the cross term ----
    for (int i = tid; i < 128 * 64; i += 256) {
      const int dk = i >> 6, dv = i & 63;       // contiguous S reads
      SbT[dv * LDQ + dk] = (__bf16)S[i];
    }
    __syncthreads();

    // ---- o = qe @ S_old + A @ v  (2 output tiles per wave) ----
    {
      const int dvt = wave >> 1;
      const int rt0 = (wave & 1) * 2;
      for (int rr = 0; rr < 2; ++rr) {
        const int rt = rt0 + rr;
        v8f acc = {};
#pragma unroll
        for (int k0 = 0; k0 < HQK; k0 += 32) {          // cross: K = 128
          v16bf a  = frag_a_rm(qe + rt * 16 * LDQ + k0, LDQ, lane);
          v16bf bb = frag_b_nk(SbT + dvt * 16 * LDQ + k0, LDQ, lane);
          acc = wmma_bf16(a, bb, acc);
        }
#pragma unroll
        for (int k0 = 0; k0 < CHUNK; k0 += 32) {        // intra: K = 64
          v16bf a  = frag_a_rm(Ab + rt * 16 * LDT + k0, LDT, lane);
          v16bf bb = frag_b_nk(vbT + dvt * 16 * LDT + k0, LDT, lane);
          acc = wmma_bf16(a, bb, acc);
        }
#pragma unroll
        for (int r = 0; r < 8; ++r) {
          const int t = rt * 16 + r + ((lane >> 4) << 3);
          const int row = dir ? (NSEQ - 1 - (cc * CHUNK + t)) : (cc * CHUNK + t);
          od[((size_t)((b * NH + h) * NSEQ) + row) * HV +
             dvb * 64 + dvt * 16 + (lane & 15)] = acc[r];
        }
      }
    }

    // ---- S = diag(exp(btot)) * S + kt^T @ v  (4 state tiles per wave) ----
    {
      const int dvt = wave >> 1;
      const int dk0 = (wave & 1) * 4;
      for (int dd = 0; dd < 4; ++dd) {
        const int dkt = dk0 + dd;
        v8f acc = {};
#pragma unroll
        for (int k0 = 0; k0 < CHUNK; k0 += 32) {
          v16bf a  = frag_a_rm(ktT + dkt * 16 * LDT + k0, LDT, lane);
          v16bf bb = frag_b_nk(vbT + dvt * 16 * LDT + k0, LDT, lane);
          acc = wmma_bf16(a, bb, acc);
        }
#pragma unroll
        for (int r = 0; r < 8; ++r) {
          const int drow = dkt * 16 + r + ((lane >> 4) << 3);
          const int dcol = dvt * 16 + (lane & 15);
          const int idx = drow * 64 + dcol;
          S[idx] = S[idx] * ebt[drow] + acc[r];
        }
      }
    }
    __syncthreads();
  }
}

// ------------------- RMS norms + SiLU gate fusion --------------------------

__global__ __launch_bounds__(256)
void gate_norm_kernel(const float* __restrict__ o_f, const float* __restrict__ o_b,
                      const float* __restrict__ g_raw, const float* __restrict__ g_b,
                      const float* __restrict__ gnorm_w, const float* __restrict__ lnorm_w,
                      __bf16* __restrict__ og) {
  __shared__ float red[256];
  const int m = blockIdx.x;
  const int b = m / NSEQ, n = m % NSEQ;
  const int tid = threadIdx.x;
  float outv[NH];
#pragma unroll
  for (int h = 0; h < NH; ++h) {
    const size_t base = ((size_t)((b * NH + h) * NSEQ) + n) * HV + tid;
    const float vf = o_f[base];
    const float vr = o_b[base];
    red[tid] = vf * vf; __syncthreads();
    for (int s = 128; s > 0; s >>= 1) { if (tid < s) red[tid] += red[tid + s]; __syncthreads(); }
    const float rsf = rsqrtf(red[0] * (1.f / HV) + 1e-5f); __syncthreads();
    red[tid] = vr * vr; __syncthreads();
    for (int s = 128; s > 0; s >>= 1) { if (tid < s) red[tid] += red[tid + s]; __syncthreads(); }
    const float rsb = rsqrtf(red[0] * (1.f / HV) + 1e-5f); __syncthreads();
    outv[h] = vf * rsf * gnorm_w[tid] + vr * rsb * lnorm_w[tid];
  }
#pragma unroll
  for (int h = 0; h < NH; ++h) {
    const int c = h * HV + tid;
    const float g = g_raw[(size_t)m * VALD + c] + g_b[c];
    const float sg = g / (1.f + __expf(-g));
    og[(size_t)m * VALD + c] = (__bf16)(sg * outv[h]);
  }
}

// ------------------------------- launcher ----------------------------------

extern "C" void kernel_launch(void* const* d_in, const int* in_sizes, int n_in,
                              void* d_out, int out_size, void* d_ws, size_t ws_size,
                              hipStream_t stream) {
  const float* x       = (const float*)d_in[0];
  const float* conv_w  = (const float*)d_in[1];
  const float* qkv_w   = (const float*)d_in[2];
  const float* gk_w1   = (const float*)d_in[3];
  const float* gk_w2   = (const float*)d_in[4];
  const float* gk_b2   = (const float*)d_in[5];
  const float* g_w     = (const float*)d_in[6];
  const float* g_b     = (const float*)d_in[7];
  const float* gnorm_w = (const float*)d_in[8];
  const float* lnorm_w = (const float*)d_in[9];
  const float* o_w     = (const float*)d_in[10];
  (void)in_sizes; (void)n_in; (void)out_size; (void)ws_size;

  char* ws = (char*)d_ws;
  size_t off = 0;
  auto alloc = [&](size_t bytes) {
    char* p = ws + off;
    off = (off + bytes + 255) & ~(size_t)255;
    return p;
  };
  __bf16* W1    = (__bf16*)alloc((size_t)NFUSE * 512 * 2);
  __bf16* W2    = (__bf16*)alloc((size_t)512 * 1024 * 2);
  __bf16* xsb   = (__bf16*)alloc((size_t)MROWS * 512 * 2);
  __bf16* qkvb  = (__bf16*)alloc((size_t)MROWS * 2048 * 2);
  float*  g_raw = (float*)alloc((size_t)MROWS * 1024 * 4);
  float*  gk1   = (float*)alloc((size_t)MROWS * 16 * 4);
  float*  gkb   = (float*)alloc((size_t)MROWS * 1024 * 4);
  float*  o_f   = (float*)alloc((size_t)MROWS * 1024 * 4);
  float*  o_bk  = (float*)alloc((size_t)MROWS * 1024 * 4);
  __bf16* og    = (__bf16*)alloc((size_t)MROWS * 1024 * 2);

  // LDS bytes for the GLA scan (all segments 16B aligned):
  // S 32768 + SbT 17408 + gbuf 32768 + qe 17408 + kd 17408
  // + ktT 18432 + vbT 9216 + Ab 9216 + btot/ebt 1024 = 155648
  const size_t gla_lds = 155648;

  // 1) pack weights to bf16
  pack_weights_kernel<<<8224, 256, 0, stream>>>(qkv_w, g_w, gk_w1, o_w, W1, W2);
  // 2) depthwise conv + SiLU -> xs (bf16)
  conv_silu_kernel<<<50176, 256, 0, stream>>>(x, conv_w, xsb);
  // 3) fused projection GEMM: [25088,512] @ [3088,512]^T (block tile 64x256)
  wmma_gemm_kernel<0><<<dim3(13, 392), 256, 0, stream>>>(
      xsb, W1, MROWS, NFUSE, 512, qkvb, g_raw, gk1, nullptr);
  // 4) low-rank gate expansion + log-sigmoid
  gk_kernel<<<100352, 256, 0, stream>>>(gk1, gk_w2, gk_b2, gkb);
  // 5) bidirectional chunked GLA scan (LDS-resident f32 state)
  gla_kernel<<<256, 256, gla_lds, stream>>>(qkvb, gkb, o_f, o_bk);
  // 6) RMS norms + SiLU gate
  gate_norm_kernel<<<MROWS, 256, 0, stream>>>(o_f, o_bk, g_raw, g_b,
                                              gnorm_w, lnorm_w, og);
  // 7) output projection: [25088,1024] @ [512,1024]^T -> d_out (f32)
  wmma_gemm_kernel<1><<<dim3(2, 392), 256, 0, stream>>>(
      og, W2, MROWS, 512, 1024, nullptr, nullptr, nullptr, (float*)d_out);
}
